// BertSumAttention_5231270167030
// MI455X (gfx1250) — matrix operand order repaired and checked by reading, not compiled, not run on hardware
//
#include <hip/hip_runtime.h>
#include <hip/hip_bf16.h>

// ---------------------------------------------------------------------------
// BERT self-attention for MI455X (gfx1250, wave32, WMMA + TDM).
//   1) gemm_wmma<float,0>   : Q/K/V projections (f32 in, f16 head-layout out)
//   2) attn_kernel          : flash attention; Q/K tiles staged by the Tensor
//                             Data Mover (double-buffered K, TENSORcnt sync);
//                             mask staged byte-swizzled through LDS.
//   3) gemm_wmma<_Float16,1>: output projection (f16 ctx in, f32 out)
// ---------------------------------------------------------------------------

typedef __attribute__((ext_vector_type(16))) _Float16 v16h;
typedef __attribute__((ext_vector_type(8)))  _Float16 v8h;
typedef __attribute__((ext_vector_type(4)))  _Float16 v4h;
typedef __attribute__((ext_vector_type(8)))  float    v8f;
typedef __attribute__((ext_vector_type(4)))  float    v4f;
typedef __attribute__((ext_vector_type(4)))  unsigned int v4u;
typedef __attribute__((ext_vector_type(8)))  int      v8i;
typedef __attribute__((ext_vector_type(4)))  int      v4i;

static constexpr int Bc  = 4;
static constexpr int Sc  = 2048;
static constexpr int Dc  = 1024;
static constexpr int Hc  = 16;
static constexpr int DHc = 64;

union F16x16 { v16h h; v4f f4[2]; };

// A operand 16x32 f16 (M x K): lane 0-15 -> M=lane, K {klo..klo+7, klo+16..klo+23}.
__device__ inline v16h frag_a(const _Float16* tile, int ld) {
  const int lane = threadIdx.x & 31;
  const int m    = lane & 15;
  const int klo  = (lane >> 4) << 3;            // 0 or 8
  const _Float16* r = tile + m * ld;
  F16x16 u;
  u.f4[0] = *(const v4f*)(r + klo);
  u.f4[1] = *(const v4f*)(r + klo + 16);
  return u.h;
}

// B operand 32x16 f16 (K x N) stored as [N][K] rows: lane -> N=lane&15,
// K {klo..klo+7, klo+16..klo+23} with klo = 0 / 8.
__device__ inline v16h frag_b(const _Float16* tile, int ld) {
  const int lane = threadIdx.x & 31;
  const int n    = lane & 15;
  const int klo  = (lane >> 4) << 3;            // 0 or 8
  const _Float16* r = tile + n * ld + klo;
  F16x16 u;
  u.f4[0] = *(const v4f*)(r);
  u.f4[1] = *(const v4f*)(r + 16);
  return u.h;
}

__device__ inline v8f wmma16(v16h a, v16h b, v8f c) {
  return __builtin_amdgcn_wmma_f32_16x16x32_f16(false, a, false, b,
                                                (short)0, c, false, false);
}

__device__ inline v8f vzero8() { v8f z = {0.f,0.f,0.f,0.f,0.f,0.f,0.f,0.f}; return z; }

// ---------------------------------------------------------------------------
// TDM: 2D f16 tile (64 x rows) global -> LDS, LDS row stride 72 halves via
// pad_enable (pad_interval=4 -> 32 DWORDs = 64 halves, pad_amount=3 -> 4 DWORDs
// = 8 halves). D# layout per CDNA5 ISA ch.10 (groups 0/1; groups 2/3 zero = 2D).
// This toolchain exposes the 6-arg builtin: (g0, g1, g2, g3, g_extra, cpol).
// ---------------------------------------------------------------------------
__device__ inline void tdm_load_tile_f16(const _Float16* gsrc, unsigned lds_byte,
                                         int rows, int row_stride_elems,
                                         int tensor_rows) {
  const unsigned long long ga = (unsigned long long)(size_t)gsrc;
  v4u g0;
  g0[0] = 1u;                                            // count=1, no gather
  g0[1] = lds_byte;                                      // lds_addr
  g0[2] = (unsigned)ga;                                  // global_addr[31:0]
  g0[3] = (unsigned)((ga >> 32) & 0x01FFFFFFu) | (2u << 30); // addr[56:32], type=2
  v8i g1;
  g1[0] = (int)((1u << 16)      // data_size = 2 bytes
              | (1u << 20)      // pad_enable
              | (4u << 22)      // pad_interval: 32 DWORDs (64 halves)
              | (3u << 25));    // pad_amount : 4 DWORDs (8 halves)
  g1[1] = (int)(64u << 16);                              // tensor_dim0 = 64
  g1[2] = (int)(((unsigned)tensor_rows & 0xffffu) << 16);// tensor_dim1 lo16
  g1[3] = (int)(64u << 16);                              // tile_dim0 = 64
  g1[4] = (int)(rows & 0xffff);                          // tile_dim1; tile_dim2=0
  g1[5] = row_stride_elems;                              // tensor_dim0_stride
  g1[6] = 0;
  g1[7] = 0;
  const v4i z4 = {0, 0, 0, 0};
  const v8i z8 = {0, 0, 0, 0, 0, 0, 0, 0};
  __builtin_amdgcn_tensor_load_to_lds(g0, g1, z4, z4, z8, 0);
}

__device__ inline unsigned lds_addr_of(const void* p) {
  return (unsigned)(unsigned long long)(size_t)p;        // addr[31:0] = LDS offset
}

// ---------------------------------------------------------------------------
// GEMM: out[m,n] = sum_k X[m,k]*W[n,k] + bias[n]   (M=8192, N=1024, K=1024)
// 128x128 tile / 8 waves, K-step 64 (16 WMMAs per wave per iteration).
// MODE 0: f16 head-layout out, (acc+bias)*scale.  MODE 1: f32 row-major out.
// ---------------------------------------------------------------------------
static constexpr int GLD = 72;   // LDS row stride in halves (64 + 8 pad, 144B)

template<typename TX, int MODE>
__global__ __launch_bounds__(256)
void gemm_wmma(const TX* __restrict__ X, const float* __restrict__ W,
               const float* __restrict__ bias, _Float16* __restrict__ outH,
               float* __restrict__ outF, float scale) {
  __shared__ _Float16 Xs[128 * GLD];
  __shared__ _Float16 Ws[128 * GLD];

  const int tid   = threadIdx.x;
  const int lane  = tid & 31;
  const int wave  = tid >> 5;
  const int m0    = blockIdx.y * 128;
  const int n0    = blockIdx.x * 128;
  const int waveM = (wave >> 2) * 64;
  const int waveN = (wave & 3) * 32;

  v8f acc[4][2];
#pragma unroll
  for (int i = 0; i < 4; ++i)
#pragma unroll
    for (int j = 0; j < 2; ++j) acc[i][j] = vzero8();

  for (int kk = 0; kk < Dc; kk += 64) {
    __syncthreads();
    // stage X[128x64] and W[128x64] tiles into LDS as f16 (16B global loads)
#pragma unroll
    for (int u = tid; u < 128 * 16; u += 256) {
      const int row = u >> 4;
      const int cg  = (u & 15) * 4;
      if constexpr (sizeof(TX) == 4) {
        v4f xv = *(const v4f*)(X + (size_t)(m0 + row) * Dc + kk + cg);
        _Float16* d = Xs + row * GLD + cg;
        d[0] = (_Float16)xv[0]; d[1] = (_Float16)xv[1];
        d[2] = (_Float16)xv[2]; d[3] = (_Float16)xv[3];
      } else {
        *(v4h*)(Xs + row * GLD + cg) =
            *(const v4h*)((const _Float16*)X + (size_t)(m0 + row) * Dc + kk + cg);
      }
      v4f wv = *(const v4f*)(W + (size_t)(n0 + row) * Dc + kk + cg);
      _Float16* dw = Ws + row * GLD + cg;
      dw[0] = (_Float16)wv[0]; dw[1] = (_Float16)wv[1];
      dw[2] = (_Float16)wv[2]; dw[3] = (_Float16)wv[3];
    }
    __syncthreads();

#pragma unroll
    for (int kd = 0; kd < 64; kd += 32) {
      v16h bf[2];
#pragma unroll
      for (int tj = 0; tj < 2; ++tj)
        bf[tj] = frag_b(Ws + (waveN + tj * 16) * GLD + kd, GLD);
#pragma unroll
      for (int ti = 0; ti < 4; ++ti) {
        v16h af = frag_a(Xs + (waveM + ti * 16) * GLD + kd, GLD);
#pragma unroll
        for (int tj = 0; tj < 2; ++tj)
          acc[ti][tj] = wmma16(af, bf[tj], acc[ti][tj]);
      }
    }
  }

  const int half = lane >> 4;
  const int colL = lane & 15;
#pragma unroll
  for (int ti = 0; ti < 4; ++ti)
#pragma unroll
    for (int tj = 0; tj < 2; ++tj)
#pragma unroll
      for (int r = 0; r < 8; ++r) {
        const int m = m0 + waveM + ti * 16 + r + half * 8;
        const int n = n0 + waveN + tj * 16 + colL;
        float v = acc[ti][tj][r] + bias[n];
        if constexpr (MODE == 0) {
          v *= scale;
          const int bb = m >> 11, s = m & (Sc - 1);
          const int hh = n >> 6,  dd = n & (DHc - 1);
          outH[(((size_t)(bb * Hc + hh)) * Sc + s) * DHc + dd] = (_Float16)v;
        } else {
          outF[(size_t)m * Dc + n] = v;
        }
      }
}

// ---------------------------------------------------------------------------
// Flash attention: block = (b, h, 64 queries), 4 waves. Q + double-buffered K
// tiles staged by TDM (wave 0 issues, TENSORcnt-synced, next-K prefetched
// during compute); V staged via VGPRs (transposed); mask tile staged through
// LDS with a byte swizzle so each lane's 4 fragment bytes are one dword.
// ---------------------------------------------------------------------------
static constexpr int LDT = 72;

__global__ __launch_bounds__(128)
void attn_kernel(const _Float16* __restrict__ Qh, const _Float16* __restrict__ Kh,
                 const _Float16* __restrict__ Vh, const unsigned char* __restrict__ mask,
                 _Float16* __restrict__ Ctx) {
  __shared__ _Float16 Qs[64 * LDT];          // [q][dh]       (TDM)
  __shared__ _Float16 Ks[2][64 * LDT];       // [key][dh] x2  (TDM, double buffer)
  __shared__ _Float16 Vt[64 * LDT];          // [dh][key]     (VGPR transpose)
  __shared__ _Float16 Ps[4 * 16 * LDT];      // per-wave P tiles
  __shared__ unsigned char Ms[64 * 64];      // mask swizzle: [q][col*4 + j]

  const int tid  = threadIdx.x;
  const int lane = tid & 31;
  const int wave = tid >> 5;
  const int qb   = blockIdx.x * 64;
  const int h    = blockIdx.y;
  const int b    = blockIdx.z;

  const size_t bh = ((size_t)b * Hc + h) * Sc;
  const _Float16* Qg = Qh + (bh + qb) * DHc;
  const _Float16* Kg = Kh + bh * DHc;
  const _Float16* Vg = Vh + bh * DHc;
  const unsigned char* mrow = mask + ((size_t)b * Sc + qb) * Sc;

  // prologue: TDM Q tile + first K tile (wave 0 only; EXEC-independent DMA)
  if (wave == 0) {
    tdm_load_tile_f16(Qg, lds_addr_of(Qs), 64, DHc, Sc);
    tdm_load_tile_f16(Kg, lds_addr_of(Ks[0]), 64, DHc, Sc);
  }

  v8f O[4];
#pragma unroll
  for (int t = 0; t < 4; ++t) O[t] = vzero8();
  float Mrow[8], Lrow[8];
#pragma unroll
  for (int r = 0; r < 8; ++r) { Mrow[r] = -INFINITY; Lrow[r] = 0.f; }

  const int half = lane >> 4;
  const int col  = lane & 15;
  const int qloc = wave * 16;

  for (int kb = 0; kb < Sc; kb += 64) {
    const int buf = (kb >> 6) & 1;
    __syncthreads();   // previous compute done: Vt/Ms and Ks[buf^1] reusable

    // stage V tile transposed (VGPR path: TDM cannot transpose)
#pragma unroll
    for (int u = tid; u < 64 * 8; u += 128) {
      const int r = u >> 3, c = (u & 7) * 8;
      v8h vv = *(const v8h*)(Vg + (size_t)(kb + r) * DHc + c);
#pragma unroll
      for (int i = 0; i < 8; ++i) Vt[(c + i) * LDT + r] = vv[i];
    }

    // stage 64x64 mask tile byte-swizzled: Ms[row][ (k&15)*4 + (k>>4) ].
    // A lane later reads its 4 fragment bytes (k = col + j*16) as ONE dword.
#pragma unroll
    for (int u = tid; u < 256; u += 128) {         // 256 chunks of 16 bytes
      const int row = u >> 2;
      const int c16 = (u & 3) * 16;
      v4u mv = *(const v4u*)(mrow + (size_t)row * Sc + kb + c16);
      unsigned char* dst = Ms + row * 64 + (c16 >> 4);
#pragma unroll
      for (int i = 0; i < 16; ++i)
        dst[i * 4] = (unsigned char)(mv[i >> 2] >> ((i & 3) * 8));
    }

    // prefetch next K tile; ensure current K tile landed (TDM is in-order)
    if (wave == 0) {
      if (kb + 64 < Sc) {
        tdm_load_tile_f16(Kg + (size_t)(kb + 64) * DHc,
                          lds_addr_of(Ks[buf ^ 1]), 64, DHc, Sc);
        __builtin_amdgcn_s_wait_tensorcnt(1);   // newest may fly; older complete
      } else {
        __builtin_amdgcn_s_wait_tensorcnt(0);
      }
    }
    __syncthreads();   // Vt/Ms staged + Ks[buf] ready for all waves

    // scores: S(16q x 64k) = Q(16x64) @ K^T, two K=32 dh-steps
    v8f sc[4];
#pragma unroll
    for (int j = 0; j < 4; ++j) sc[j] = vzero8();
#pragma unroll
    for (int kd = 0; kd < DHc; kd += 32) {
      v16h aq = frag_a(Qs + qloc * LDT + kd, LDT);
#pragma unroll
      for (int j = 0; j < 4; ++j) {
        v16h bk = frag_b(Ks[buf] + (j * 16) * LDT + kd, LDT);
        sc[j] = wmma16(aq, bk, sc[j]);
      }
    }

    // mask + online softmax (fully unrolled; one ds_load_b32 per row)
#pragma unroll
    for (int r = 0; r < 8; ++r) {
      const unsigned mw =
          *(const unsigned*)(Ms + (qloc + r + half * 8) * 64 + col * 4);
      float s[4];
      float smax = -INFINITY;
#pragma unroll
      for (int j = 0; j < 4; ++j) {
        float v = sc[j][r];
        if ((mw >> (j * 8)) & 0xffu) v = -1e18f;
        s[j] = v;
        smax = fmaxf(smax, v);
      }
#pragma unroll
      for (int off = 8; off; off >>= 1)
        smax = fmaxf(smax, __shfl_xor(smax, off, 32));
      const float mold = Mrow[r];
      const float mnew = fmaxf(mold, smax);
      const float corr = __expf(mold - mnew);
      Mrow[r] = mnew;
      float ps = 0.f;
#pragma unroll
      for (int j = 0; j < 4; ++j) {
        const float p = __expf(s[j] - mnew);
        ps += p;
        Ps[(wave * 16 + r + half * 8) * LDT + j * 16 + col] = (_Float16)p;
      }
#pragma unroll
      for (int off = 8; off; off >>= 1)
        ps += __shfl_xor(ps, off, 32);
      Lrow[r] = Lrow[r] * corr + ps;
#pragma unroll
      for (int t = 0; t < 4; ++t) O[t][r] *= corr;
    }

    // O += P(16x64) @ V(64x64) (per-wave LDS P; DS in-order within wave)
#pragma unroll
    for (int ks = 0; ks < 64; ks += 32) {
      v16h ap = frag_a(Ps + (wave * 16) * LDT + ks, LDT);
#pragma unroll
      for (int t = 0; t < 4; ++t) {
        v16h bv = frag_b(Vt + (t * 16) * LDT + ks, LDT);
        O[t] = wmma16(ap, bv, O[t]);
      }
    }
  }

  // ctx[b*S + q][h*64 + dh] = O / L  (f16)
  _Float16* cg = Ctx + ((size_t)(b * Sc + qb + qloc)) * Dc + h * DHc;
#pragma unroll
  for (int t = 0; t < 4; ++t)
#pragma unroll
    for (int r = 0; r < 8; ++r) {
      const float v = O[t][r] / Lrow[r];
      cg[(size_t)(r + half * 8) * Dc + t * 16 + col] = (_Float16)v;
    }
}

// ---------------------------------------------------------------------------
extern "C" void kernel_launch(void* const* d_in, const int* in_sizes, int n_in,
                              void* d_out, int out_size, void* d_ws, size_t ws_size,
                              hipStream_t stream) {
  const float*         data = (const float*)d_in[0];
  const unsigned char* mask = (const unsigned char*)d_in[1];  // jnp.bool_: 1 B/elem
  const float* Wq = (const float*)d_in[2];
  const float* bq = (const float*)d_in[3];
  const float* Wk = (const float*)d_in[4];
  const float* bk = (const float*)d_in[5];
  const float* Wv = (const float*)d_in[6];
  const float* bv = (const float*)d_in[7];
  const float* Wo = (const float*)d_in[8];
  const float* bo = (const float*)d_in[9];

  const size_t BSD = (size_t)Bc * Sc * Dc;
  _Float16* Qh  = (_Float16*)d_ws;
  _Float16* Kh  = Qh + BSD;
  _Float16* Vh  = Kh + BSD;
  _Float16* Ctx = Vh + BSD;                  // 64 MB f16 workspace total

  const dim3 gGemm(Dc / 128, (Bc * Sc) / 128);   // (8, 64)
  const dim3 bGemm(256);
  const float qscale = 0.125f;                   // 1/sqrt(DH)

  gemm_wmma<float, 0><<<gGemm, bGemm, 0, stream>>>(data, Wq, bq, Qh, nullptr, qscale);
  gemm_wmma<float, 0><<<gGemm, bGemm, 0, stream>>>(data, Wk, bk, Kh, nullptr, 1.0f);
  gemm_wmma<float, 0><<<gGemm, bGemm, 0, stream>>>(data, Wv, bv, Vh, nullptr, 1.0f);

  attn_kernel<<<dim3(Sc / 64, Hc, Bc), dim3(128), 0, stream>>>(Qh, Kh, Vh, mask, Ctx);

  gemm_wmma<_Float16, 1><<<gGemm, bGemm, 0, stream>>>(Ctx, Wo, bo, nullptr,
                                                      (float*)d_out, 1.0f);
}
